// Encoder_37469294691188
// MI455X (gfx1250) — compile-verified, matrix operand order checked by reading, not run on hardware
//
#include <hip/hip_runtime.h>

// ---------------------------------------------------------------------------
// CDNA5 (gfx1250) transformer encoder: f16 WMMA GEMMs + flash attention.
// ---------------------------------------------------------------------------

typedef __attribute__((ext_vector_type(16))) _Float16 v16h;
typedef __attribute__((ext_vector_type(8)))  _Float16 v8h;
typedef __attribute__((ext_vector_type(2)))  _Float16 v2h;
typedef __attribute__((ext_vector_type(8)))  float    v8f;

#define DEV static __device__ __forceinline__

constexpr int B_  = 8;
constexpr int S_  = 1024;
constexpr int D_  = 1024;
constexpr int H_  = 16;
constexpr int F_  = 4096;
constexpr int L_  = 6;
constexpr int HD_ = 64;
constexpr int M_  = B_ * S_;   // 8192 rows

DEV v8f vzero8() {
  v8f z;
#pragma unroll
  for (int i = 0; i < 8; ++i) z[i] = 0.0f;
  return z;
}

DEV v8f wmma_f16(v16h a, v16h b, v8f c) {
  // D = A(16x32 f16) x B(32x16 f16) + C(16x16 f32)
  return __builtin_amdgcn_wmma_f32_16x16x32_f16(
      /*neg_a=*/false, a, /*neg_b=*/false, b,
      /*c_mod=*/(short)0, c, /*reuse_a=*/false, /*reuse_b=*/false);
}

// ---------------------------------------------------------------------------
// Embedding * sqrt(D) + sinusoidal PE
// ---------------------------------------------------------------------------
__global__ __launch_bounds__(256) void embed_kernel(
    const int* __restrict__ src, const float* __restrict__ emb,
    float* __restrict__ x) {
  const int row  = blockIdx.x;           // b*S + s
  const int spos = row & (S_ - 1);
  const int tid  = threadIdx.x;
  const int tok  = src[row];
#pragma unroll
  for (int j = 0; j < 4; ++j) {
    const int d = tid + j * 256;
    const float freq = __expf(-(float)(2 * (d >> 1)) *
                              (9.210340371976184f / (float)D_)); // ln(10000)/D
    const float ang = (float)spos * freq;
    const float pe  = (d & 1) ? __cosf(ang) : __sinf(ang);
    x[(size_t)row * D_ + d] = emb[(size_t)tok * D_ + d] * 32.0f + pe; // sqrt(1024)=32
  }
}

// ---------------------------------------------------------------------------
// LayerNorm over last dim (D=1024), one block per row
// ---------------------------------------------------------------------------
__global__ __launch_bounds__(256) void ln_kernel(
    const float* __restrict__ x, const float* __restrict__ g,
    const float* __restrict__ bta, float* __restrict__ out) {
  __shared__ float s1[256];
  __shared__ float s2[256];
  const int row = blockIdx.x;
  const int tid = threadIdx.x;
  const float* xr = x + (size_t)row * D_;
  float vals[4], s = 0.0f, q = 0.0f;
#pragma unroll
  for (int j = 0; j < 4; ++j) {
    vals[j] = xr[tid + j * 256];
    s += vals[j];
    q += vals[j] * vals[j];
  }
  s1[tid] = s;
  s2[tid] = q;
  __syncthreads();
  for (int off = 128; off > 0; off >>= 1) {
    if (tid < off) {
      s1[tid] += s1[tid + off];
      s2[tid] += s2[tid + off];
    }
    __syncthreads();
  }
  const float mu  = s1[0] * (1.0f / (float)D_);
  const float var = s2[0] * (1.0f / (float)D_) - mu * mu;
  const float rs  = rsqrtf(var + 1e-5f);
#pragma unroll
  for (int j = 0; j < 4; ++j) {
    const int d = tid + j * 256;
    out[(size_t)row * D_ + d] = (vals[j] - mu) * rs * g[d] + bta[d];
  }
}

// ---------------------------------------------------------------------------
// Tiled WMMA GEMM: C[M,N] = act(A[M,K] x W[K,N] + bias (+ residual))
// Block tile 128x128, 8 waves (4x2), wave tile 32x64, K-step 64 (2 WMMA steps).
// Epilogue behavior is compile-time (no per-element branches).
// ---------------------------------------------------------------------------
template <bool HAS_RES, bool RELU>
__global__ __launch_bounds__(256) void gemm_kernel(
    const float* __restrict__ A, const float* __restrict__ W,
    const float* __restrict__ bias, const float* __restrict__ residual,
    float* __restrict__ C, int M, int N, int K) {
  constexpr int LDA = 72;  // 64 halves + pad (144B rows, 16B-aligned frags)
  constexpr int LDB = 72;
  __shared__ __align__(16) _Float16 As[128 * LDA];  // [m][k]
  __shared__ __align__(16) _Float16 Bs[128 * LDB];  // [n][k] (transposed)

  const int tid   = threadIdx.x;
  const int lane  = tid & 31;
  const int wid   = tid >> 5;
  const int hi    = lane >> 4;
  const int khalf = hi * 8;
  const int wm    = (wid & 3) * 32;   // wave m-offset in tile
  const int wn    = (wid >> 2) * 64;  // wave n-offset in tile
  const int m0    = blockIdx.y * 128;
  const int n0    = blockIdx.x * 128;

  v8f acc[2][4];
#pragma unroll
  for (int i = 0; i < 2; ++i)
#pragma unroll
    for (int j = 0; j < 4; ++j) acc[i][j] = vzero8();

  const int ktiles = K >> 6;
  for (int kt = 0; kt < ktiles; ++kt) {
    // ---- batched global loads for this K-tile --------------------------
    float4 fa[8];
#pragma unroll
    for (int it = 0; it < 8; ++it) {
      const int id = tid + it * 256;
      const int r  = id >> 4;         // m row (0..127)
      const int c4 = (id & 15) * 4;   // k col
      fa[it] = *(const float4*)(A + (size_t)(m0 + r) * K + kt * 64 + c4);
    }
    float4 fb[8];
#pragma unroll
    for (int it = 0; it < 4; ++it) {
      const int id = tid + it * 256;
      const int rp = id >> 5;         // k row pair (0..31)
      const int c4 = (id & 31) * 4;   // n col
      const float* base = W + (size_t)(kt * 64 + 2 * rp) * N + n0 + c4;
      fb[2 * it]     = *(const float4*)(base);
      fb[2 * it + 1] = *(const float4*)(base + N);
    }
    // Speculative prefetch of next K-tile (safe past end: dropped on fault).
    {
      const int r = tid >> 1;
      __builtin_prefetch(A + (size_t)(m0 + r) * K + (kt + 1) * 64 +
                             (tid & 1) * 32, 0, 1);
      const int rr = tid >> 2;
      __builtin_prefetch(W + (size_t)((kt + 1) * 64 + rr) * N + n0 +
                             (tid & 3) * 32, 0, 1);
    }

    __syncthreads();  // previous iteration's LDS reads done
    // ---- convert fp32 -> f16 and store to LDS --------------------------
#pragma unroll
    for (int it = 0; it < 8; ++it) {
      const int id = tid + it * 256;
      const int r  = id >> 4;
      const int c4 = (id & 15) * 4;
      _Float16* p = &As[r * LDA + c4];
      p[0] = (_Float16)fa[it].x; p[1] = (_Float16)fa[it].y;
      p[2] = (_Float16)fa[it].z; p[3] = (_Float16)fa[it].w;
    }
#pragma unroll
    for (int it = 0; it < 4; ++it) {
      const int id = tid + it * 256;
      const int rp = id >> 5;
      const int c4 = (id & 31) * 4;
      const float a0[4] = {fb[2 * it].x, fb[2 * it].y, fb[2 * it].z,
                           fb[2 * it].w};
      const float a1[4] = {fb[2 * it + 1].x, fb[2 * it + 1].y,
                           fb[2 * it + 1].z, fb[2 * it + 1].w};
#pragma unroll
      for (int j = 0; j < 4; ++j) {
        v2h t;
        t[0] = (_Float16)a0[j];
        t[1] = (_Float16)a1[j];
        *(v2h*)&Bs[(c4 + j) * LDB + 2 * rp] = t;  // packed ds_store_b32
      }
    }
    __syncthreads();

    // ---- two 16x16x32 WMMA sub-steps over the 64-wide K tile -----------
#pragma unroll
    for (int c = 0; c < 2; ++c) {
      v16h af[2], bf[4];
#pragma unroll
      for (int mt = 0; mt < 2; ++mt) {
        const _Float16* p =
            &As[(wm + mt * 16 + (lane & 15)) * LDA + c * 32 + khalf];
        *(v8h*)&af[mt]       = *(const v8h*)p;
        *((v8h*)&af[mt] + 1) = *(const v8h*)(p + 16);
      }
#pragma unroll
      for (int nt = 0; nt < 4; ++nt) {
        const _Float16* p =
            &Bs[(wn + nt * 16 + (lane & 15)) * LDB + c * 32 + khalf];
        *(v8h*)&bf[nt]       = *(const v8h*)p;
        *((v8h*)&bf[nt] + 1) = *(const v8h*)(p + 16);
      }
#pragma unroll
      for (int mt = 0; mt < 2; ++mt)
#pragma unroll
        for (int nt = 0; nt < 4; ++nt)
          acc[mt][nt] = wmma_f16(af[mt], bf[nt], acc[mt][nt]);
    }
  }

  // Epilogue: C layout => n = lane&15, m = v + 8*(lane>>4) within 16x16 tile.
#pragma unroll
  for (int mt = 0; mt < 2; ++mt) {
#pragma unroll
    for (int nt = 0; nt < 4; ++nt) {
      const int n  = n0 + wn + nt * 16 + (lane & 15);
      const float bv = bias[n];
      float res[8];
      if (HAS_RES) {
#pragma unroll
        for (int v = 0; v < 8; ++v) {
          const int m = m0 + wm + mt * 16 + v + 8 * hi;
          res[v] = residual[(size_t)m * N + n];
        }
      }
#pragma unroll
      for (int v = 0; v < 8; ++v) {
        const int m = m0 + wm + mt * 16 + v + 8 * hi;
        float val = acc[mt][nt][v] + bv;
        if (HAS_RES) val += res[v];
        if (RELU) val = fmaxf(val, 0.0f);
        C[(size_t)m * N + n] = val;
      }
    }
  }
}

// ---------------------------------------------------------------------------
// Flash attention: one block per (b, h, 64-query tile); 4 waves x 16 q-rows.
// QK^T and P@V on WMMA; online softmax with 16-lane shuffles.
// ---------------------------------------------------------------------------
__global__ __launch_bounds__(128) void attn_kernel(
    const float* __restrict__ q, const float* __restrict__ k,
    const float* __restrict__ v, const int* __restrict__ mask,
    float* __restrict__ ctx) {
  constexpr int LDT = 72;  // padded stride in halves (144B, 16B-aligned)
  __shared__ __align__(16) _Float16 Vt[64 * LDT];      // V transposed: [hd][key]
  __shared__ __align__(16) _Float16 Ps[4 * 16 * LDT];  // per-wave P: [m][key]

  const int tid   = threadIdx.x;
  const int lane  = tid & 31;
  const int wid   = tid >> 5;
  const int hi    = lane >> 4;
  const int khalf = hi * 8;
  const int b     = blockIdx.z;
  const int h     = blockIdx.y;
  const int q0    = blockIdx.x * 64 + wid * 16;

  const float* qb = q + (size_t)b * S_ * D_ + h * HD_;
  const float* kb = k + (size_t)b * S_ * D_ + h * HD_;
  const float* vb = v + (size_t)b * S_ * D_ + h * HD_;
  const int*   mb = mask + b * S_;

  // Load Q fragments for this wave's 16 rows, pre-scaled by 1/sqrt(HD)=0.125.
  v16h aq[2];
  {
    const float* qrow = qb + (size_t)(q0 + (lane & 15)) * D_;
#pragma unroll
    for (int c = 0; c < 2; ++c)
#pragma unroll
      for (int i = 0; i < 8; ++i) {
        const int ko = c * 32 + ((i < 4) ? 0 : 16) + khalf + 2 * (i & 3);
        const float2 f = *(const float2*)(qrow + ko);
        aq[c][2 * i]     = (_Float16)(f.x * 0.125f);
        aq[c][2 * i + 1] = (_Float16)(f.y * 0.125f);
      }
  }

  v8f o[4];
#pragma unroll
  for (int nt = 0; nt < 4; ++nt) o[nt] = vzero8();
  float rmax[8], rsum[8];
#pragma unroll
  for (int vv = 0; vv < 8; ++vv) { rmax[vv] = -1e30f; rsum[vv] = 0.0f; }

  for (int kt = 0; kt < S_; kt += 64) {
    // ---- batched global loads of V tile (64 keys x 64 dims) ------------
    float4 f0[4], f1[4];
#pragma unroll
    for (int it = 0; it < 4; ++it) {
      const int id = tid + it * 128;
      const int rp = id >> 4;         // key pair (0..31)
      const int c4 = (id & 15) * 4;   // hd col
      const float* base = vb + (size_t)(kt + 2 * rp) * D_ + c4;
      f0[it] = *(const float4*)(base);
      f1[it] = *(const float4*)(base + D_);
    }
    __syncthreads();  // previous iteration's Vt reads done
    // Store transposed, two keys packed per ds_store_b32.
#pragma unroll
    for (int it = 0; it < 4; ++it) {
      const int id = tid + it * 128;
      const int rp = id >> 4;
      const int c4 = (id & 15) * 4;
      const float a0[4] = {f0[it].x, f0[it].y, f0[it].z, f0[it].w};
      const float a1[4] = {f1[it].x, f1[it].y, f1[it].z, f1[it].w};
#pragma unroll
      for (int j = 0; j < 4; ++j) {
        v2h t;
        t[0] = (_Float16)a0[j];
        t[1] = (_Float16)a1[j];
        *(v2h*)&Vt[(c4 + j) * LDT + 2 * rp] = t;
      }
    }
    __syncthreads();

    // ---- scores: 4 tiles of 16x16 over this 64-key block ---------------
    v8f sc[4];
#pragma unroll
    for (int nt = 0; nt < 4; ++nt) {
      const int key = kt + nt * 16 + (lane & 15);
      const float* krow = kb + (size_t)key * D_;
      v16h bf[2];
#pragma unroll
      for (int c = 0; c < 2; ++c)
#pragma unroll
        for (int i = 0; i < 8; ++i) {
          const int ko = c * 32 + ((i < 4) ? 0 : 16) + khalf + 2 * (i & 3);
          const float2 f = *(const float2*)(krow + ko);
          bf[c][2 * i]     = (_Float16)f.x;
          bf[c][2 * i + 1] = (_Float16)f.y;
        }
      v8f s = vzero8();
      s = wmma_f16(aq[0], bf[0], s);
      s = wmma_f16(aq[1], bf[1], s);
      if (mb[key] == 0) {
#pragma unroll
        for (int e = 0; e < 8; ++e) s[e] += -1e10f;
      }
      sc[nt] = s;
    }

    // ---- online softmax. Row r = vv + 8*hi lives on 16 lanes of half hi.
#pragma unroll
    for (int vv = 0; vv < 8; ++vv) {
      float mx = sc[0][vv];
      mx = fmaxf(mx, sc[1][vv]);
      mx = fmaxf(mx, sc[2][vv]);
      mx = fmaxf(mx, sc[3][vv]);
      mx = fmaxf(mx, __shfl_xor(mx, 8));
      mx = fmaxf(mx, __shfl_xor(mx, 4));
      mx = fmaxf(mx, __shfl_xor(mx, 2));
      mx = fmaxf(mx, __shfl_xor(mx, 1));
      const float nm    = fmaxf(rmax[vv], mx);
      const float scale = __expf(rmax[vv] - nm);
      rmax[vv] = nm;
      float rs = 0.0f;
#pragma unroll
      for (int nt = 0; nt < 4; ++nt) {
        const float e = __expf(sc[nt][vv] - nm);
        sc[nt][vv] = e;
        rs += e;
      }
      rs += __shfl_xor(rs, 8);
      rs += __shfl_xor(rs, 4);
      rs += __shfl_xor(rs, 2);
      rs += __shfl_xor(rs, 1);
      rsum[vv] = rsum[vv] * scale + rs;
#pragma unroll
      for (int nt = 0; nt < 4; ++nt) o[nt][vv] *= scale;
    }

    // ---- write P (f16) to wave-private LDS, C-layout -> [m][key] -------
    _Float16* pw = &Ps[wid * 16 * LDT];
#pragma unroll
    for (int nt = 0; nt < 4; ++nt)
#pragma unroll
      for (int vv = 0; vv < 8; ++vv)
        pw[(vv + 8 * hi) * LDT + nt * 16 + (lane & 15)] = (_Float16)sc[nt][vv];

    // ---- o += P @ V  (K-dim = 64 keys, two 32-key WMMA steps) ----------
#pragma unroll
    for (int c = 0; c < 2; ++c) {
      v16h ap;
      const _Float16* p = &pw[(lane & 15) * LDT + c * 32 + khalf];
      *(v8h*)&ap       = *(const v8h*)p;
      *((v8h*)&ap + 1) = *(const v8h*)(p + 16);
#pragma unroll
      for (int nt = 0; nt < 4; ++nt) {
        v16h bv_;
        const _Float16* pv = &Vt[(nt * 16 + (lane & 15)) * LDT + c * 32 + khalf];
        *(v8h*)&bv_       = *(const v8h*)pv;
        *((v8h*)&bv_ + 1) = *(const v8h*)(pv + 16);
        o[nt] = wmma_f16(ap, bv_, o[nt]);
      }
    }
  }

  // Normalize and write ctx[b, q0+m, h*HD + col].
  float* cb = ctx + (size_t)b * S_ * D_ + h * HD_;
#pragma unroll
  for (int nt = 0; nt < 4; ++nt) {
    const int col = nt * 16 + (lane & 15);
#pragma unroll
    for (int vv = 0; vv < 8; ++vv) {
      const int row = q0 + vv + 8 * hi;
      cb[(size_t)row * D_ + col] = o[nt][vv] / rsum[vv];
    }
  }
}

// ---------------------------------------------------------------------------
// Host-side orchestration
// ---------------------------------------------------------------------------
extern "C" void kernel_launch(void* const* d_in, const int* in_sizes, int n_in,
                              void* d_out, int out_size, void* d_ws,
                              size_t ws_size, hipStream_t stream) {
  (void)in_sizes; (void)n_in; (void)out_size; (void)ws_size;
  const int*   src  = (const int*)d_in[0];
  const int*   mask = (const int*)d_in[1];
  const float* emb  = (const float*)d_in[2];
  const float* Wq   = (const float*)d_in[3];
  const float* bq   = (const float*)d_in[4];
  const float* Wk   = (const float*)d_in[5];
  const float* bk   = (const float*)d_in[6];
  const float* Wv   = (const float*)d_in[7];
  const float* bv   = (const float*)d_in[8];
  const float* Wo   = (const float*)d_in[9];
  const float* bo   = (const float*)d_in[10];
  const float* ln1g = (const float*)d_in[11];
  const float* ln1b = (const float*)d_in[12];
  const float* W1   = (const float*)d_in[13];
  const float* b1   = (const float*)d_in[14];
  const float* W2   = (const float*)d_in[15];
  const float* b2   = (const float*)d_in[16];
  const float* ln2g = (const float*)d_in[17];
  const float* ln2b = (const float*)d_in[18];

  float* ws = (float*)d_ws;
  const size_t MD = (size_t)M_ * D_;
  float* x  = ws;            // [0, MD)
  float* qp = ws + 1 * MD;   // q   (later aliased as pre-LN tmp t1)
  float* kp = ws + 2 * MD;   // k
  float* vp = ws + 3 * MD;   // v
  float* cp = ws + 4 * MD;   // ctx
  float* hb = ws + 1 * MD;   // FFN hidden, aliases q..ctx (4*MD == M_*F_)
  float* t2 = ws + 5 * MD;   // pre-LN2 tmp
  float* t1 = qp;            // pre-LN1 tmp (q dead after attention)

  dim3 blk(256);
  embed_kernel<<<dim3(M_), blk, 0, stream>>>(src, emb, x);

  for (int l = 0; l < L_; ++l) {
    const size_t wdd = (size_t)l * D_ * D_;
    const size_t wdf = (size_t)l * D_ * F_;
    dim3 gDD(D_ / 128, M_ / 128);

    gemm_kernel<false, false><<<gDD, blk, 0, stream>>>(
        x, Wq + wdd, bq + l * D_, nullptr, qp, M_, D_, D_);
    gemm_kernel<false, false><<<gDD, blk, 0, stream>>>(
        x, Wk + wdd, bk + l * D_, nullptr, kp, M_, D_, D_);
    gemm_kernel<false, false><<<gDD, blk, 0, stream>>>(
        x, Wv + wdd, bv + l * D_, nullptr, vp, M_, D_, D_);

    attn_kernel<<<dim3(S_ / 64, H_, B_), dim3(128), 0, stream>>>(qp, kp, vp,
                                                                 mask, cp);

    gemm_kernel<true, false><<<gDD, blk, 0, stream>>>(
        cp, Wo + wdd, bo + l * D_, x, t1, M_, D_, D_);
    ln_kernel<<<dim3(M_), blk, 0, stream>>>(t1, ln1g + l * D_, ln1b + l * D_,
                                            x);

    gemm_kernel<false, true><<<dim3(F_ / 128, M_ / 128), blk, 0, stream>>>(
        x, W1 + wdf, b1 + (size_t)l * F_, nullptr, hb, M_, F_, D_);
    gemm_kernel<true, false><<<gDD, blk, 0, stream>>>(
        hb, W2 + wdf, b2 + l * D_, x, t2, M_, D_, F_);

    float* outp = (l == L_ - 1) ? (float*)d_out : x;
    ln_kernel<<<dim3(M_), blk, 0, stream>>>(t2, ln2g + l * D_, ln2b + l * D_,
                                            outp);
  }
}
